// ContrastiveLoss_23450521436869
// MI455X (gfx1250) — matrix operand order repaired and checked by reading, not compiled, not run on hardware
//
#include <hip/hip_runtime.h>
#include <hip/hip_bf16.h>

#define N_ROWS   4096
#define D_DIM    256
#define TWO_N    8192
#define INV_TAU  10.0f   // 1/0.1

typedef _Float16 v16h_t __attribute__((ext_vector_type(16)));
typedef _Float16 v8h_t  __attribute__((ext_vector_type(8)));
typedef float    v8f_t  __attribute__((ext_vector_type(8)));
typedef unsigned int v4u_t __attribute__((ext_vector_type(4)));
typedef int      v4i_t  __attribute__((ext_vector_type(4)));
typedef int      v8i_t  __attribute__((ext_vector_type(8)));

// Use the CDNA5 Tensor Data Mover for LDS staging when the builtin exists.
#if defined(__has_builtin)
#if __has_builtin(__builtin_amdgcn_tensor_load_to_lds) && \
    __has_builtin(__builtin_amdgcn_s_wait_tensorcnt)
#define USE_TDM 1
#endif
#endif

// ---------------- workspace layout ----------------
// [0, 4 MiB)      : Z  (f16, 8192 x 256)
// [+32 KiB)       : inv norms (f32, 8192)
// [+32 KiB)       : rowlog (f32, 8192)  = log(denominator) per row
// [+16 KiB)       : pos (f32, 4096)
#define Z_OFF      0
#define INV_OFF    (TWO_N * D_DIM * 2)          // 4194304
#define ROWLOG_OFF (INV_OFF + TWO_N * 4)        // +32768
#define POS_OFF    (ROWLOG_OFF + TWO_N * 4)     // +32768

// -------- Kernel 1: L2-normalize rows, emit f16 Z + inv norms --------
__global__ void __launch_bounds__(256)
k_normalize(const float* __restrict__ x1, const float* __restrict__ x2,
            _Float16* __restrict__ z, float* __restrict__ inv) {
    const int row = blockIdx.x;              // 0..8191
    const int t   = threadIdx.x;             // 0..255 == feature index
    const float* src = (row < N_ROWS) ? (x1 + (size_t)row * D_DIM)
                                      : (x2 + (size_t)(row - N_ROWS) * D_DIM);
    float v  = src[t];
    float ss = v * v;
    ss += __shfl_xor(ss, 16, 32);
    ss += __shfl_xor(ss,  8, 32);
    ss += __shfl_xor(ss,  4, 32);
    ss += __shfl_xor(ss,  2, 32);
    ss += __shfl_xor(ss,  1, 32);
    __shared__ float wsum[8];
    __shared__ float s_inv;
    if ((t & 31) == 0) wsum[t >> 5] = ss;
    __syncthreads();
    if (t == 0) {
        float tot = 0.f;
#pragma unroll
        for (int i = 0; i < 8; ++i) tot += wsum[i];
        float iv = 1.0f / fmaxf(sqrtf(tot), 1e-12f);
        inv[row] = iv;
        s_inv = iv;
    }
    __syncthreads();
    z[(size_t)row * D_DIM + t] = (_Float16)(v * s_inv);
}

// -------- Kernel 2: positives in f32 --------
__global__ void __launch_bounds__(256)
k_pos(const float* __restrict__ x1, const float* __restrict__ x2,
      const float* __restrict__ inv, float* __restrict__ pos) {
    const int i = blockIdx.x;                // 0..4095
    const int t = threadIdx.x;
    float p = x1[(size_t)i * D_DIM + t] * x2[(size_t)i * D_DIM + t];
    p += __shfl_xor(p, 16, 32);
    p += __shfl_xor(p,  8, 32);
    p += __shfl_xor(p,  4, 32);
    p += __shfl_xor(p,  2, 32);
    p += __shfl_xor(p,  1, 32);
    __shared__ float wsum[8];
    if ((t & 31) == 0) wsum[t >> 5] = p;
    __syncthreads();
    if (t == 0) {
        float tot = 0.f;
#pragma unroll
        for (int k = 0; k < 8; ++k) tot += wsum[k];
        pos[i] = tot * inv[i] * inv[i + N_ROWS];
    }
}

// -------- Kernel 3: fused Z*Z^T (WMMA) + masked exp row-sum + log --------
// 64 blocks x 8 waves; each wave owns a 16-row strip. The block stages each
// 16-column tile of Z (16 contiguous rows == flat 8 KiB) into LDS via the
// Tensor Data Mover, shared by all 8 waves.
__global__ void __launch_bounds__(256)
k_gemm_rowsum(const _Float16* __restrict__ z, float* __restrict__ rowlog) {
    __shared__ _Float16 tile[16 * D_DIM];    // 8 KiB
    const int t    = threadIdx.x;
    const int wave = t >> 5;
    const int lane = t & 31;
    const int sub  = lane & 15;              // 0..15
    const int half = lane >> 4;              // 0/1
    const int R    = (blockIdx.x * 8 + wave) * 16;   // strip base row

    // ---- A fragments: 16 rows x 256 K, resident for the whole kernel ----
    // A k order used here (contiguous, as B): half==0 -> K[kb*32+0..15]
    //                        half==1 -> K[kb*32+16..31]
    v16h_t a[8];
    {
        const _Float16* zrow = z + (size_t)(R + sub) * D_DIM;
#pragma unroll
        for (int kb = 0; kb < 8; ++kb) {
            v8h_t lo = *(const v8h_t*)(zrow + kb * 32 + half * 16);
            v8h_t hi = *(const v8h_t*)(zrow + kb * 32 + half * 16 + 8);
            a[kb] = __builtin_shufflevector(lo, hi,
                        0,1,2,3,4,5,6,7,8,9,10,11,12,13,14,15);
        }
    }

    v8f_t rs = {0.f,0.f,0.f,0.f,0.f,0.f,0.f,0.f};

    for (int jt = 0; jt < TWO_N / 16; ++jt) {
        __syncthreads();                      // previous tile fully consumed
#ifdef USE_TDM
        if (wave == 0) {
            // 1D Tensor-DMA descriptor: copy 8 KiB (4096 x 2B) global -> LDS
            const unsigned long long ga =
                (unsigned long long)(size_t)(z + (size_t)jt * 16 * D_DIM);
            const unsigned lds_off = (unsigned)(size_t)tile;  // LDS aperture: addr[31:0]
            v4u_t g0;
            g0.x = 1u;                                   // count=1 (valid), user mode
            g0.y = lds_off;                              // bits[63:32]  = lds_addr
            g0.z = (unsigned)(ga & 0xFFFFFFFFu);         // bits[95:64]  = ga[31:0]
            g0.w = (unsigned)((ga >> 32) & 0x01FFFFFFu)  // bits[120:96] = ga[56:32]
                 | (2u << 30);                           // bits[127:126]= type 2 (image)
            // group1: data_size=2B, tensor_dim0=4096, tensor_dim1=1,
            //         tile_dim0=4096, tensor_dim0_stride=4096
            v8i_t g1 = { 0x00010000,   // [17:16]=1 -> 2-byte elements; mask=0
                         0x10000000,   // [31:16] = tensor_dim0 lo16 (0x1000)
                         0x00010000,   // [31:16] = tensor_dim1 lo16 (=1)
                         0x10000000,   // [31:16] = tile_dim0 (0x1000)
                         0,            // tile_dim1/2 = 0 (unused)
                         4096,         // tensor_dim0_stride lo32
                         0, 0 };
            v4i_t g2 = {0, 0, 0, 0};
            v4i_t g3 = {0, 0, 0, 0};
#if __clang_major__ >= 23
            v8i_t g4 = {0, 0, 0, 0, 0, 0, 0, 0};
            __builtin_amdgcn_tensor_load_to_lds(g0, g1, g2, g3, g4, 0);
#else
            __builtin_amdgcn_tensor_load_to_lds(g0, g1, g2, g3, 0);
#endif
            __builtin_amdgcn_s_wait_tensorcnt(0);
        }
#else
        {   // fallback: cooperative stage, 32 B per thread
            const float4* src = (const float4*)(z + (size_t)jt * 16 * D_DIM);
            float4* dst = (float4*)tile;
            dst[t * 2]     = src[t * 2];
            dst[t * 2 + 1] = src[t * 2 + 1];
        }
#endif
        __syncthreads();                      // tile published
        if (jt + 1 < TWO_N / 16)              // global_prefetch_b8 for next tile
            __builtin_prefetch(z + (size_t)(jt + 1) * 16 * D_DIM + t * 16, 0, 1);

        // ---- load all 8 B fragments first: 16 back-to-back ds_load_b128 ----
        // B (32x16) layout: lanes 0-15 hold K=kb*32+0..15 of column 'sub',
        // lanes 16-31 hold K=kb*32+16..31 -> contiguous 16 halves of row-major Z
        v16h_t b[8];
#pragma unroll
        for (int kb = 0; kb < 8; ++kb)
            b[kb] = *(const v16h_t*)(tile + sub * D_DIM + kb * 32 + half * 16);

        v8f_t c = {0.f,0.f,0.f,0.f,0.f,0.f,0.f,0.f};
#pragma unroll
        for (int kb = 0; kb < 8; ++kb)
            c = __builtin_amdgcn_wmma_f32_16x16x32_f16(
                    false, a[kb], false, b[kb], (short)0, c, false, false);

        // ---- epilogue: masked exp(sim/tau); diagonal contributes exp(0)=1 ----
        const int gcol = jt * 16 + sub;
#pragma unroll
        for (int r = 0; r < 8; ++r) {
            const int grow = R + half * 8 + r;  // C layout: M = r + 8*half, N = sub
            rs[r] += (grow == gcol) ? 1.0f : __expf(c[r] * INV_TAU);
        }
    }

    // reduce across the 16 lanes of each half (columns)
#pragma unroll
    for (int r = 0; r < 8; ++r) {
        float v = rs[r];
        v += __shfl_xor(v, 1, 32);
        v += __shfl_xor(v, 2, 32);
        v += __shfl_xor(v, 4, 32);
        v += __shfl_xor(v, 8, 32);
        rs[r] = v;
    }
    if (sub == 0) {                           // lanes 0/16 own rows R..R+7 / R+8..R+15
#pragma unroll
        for (int r = 0; r < 8; ++r)
            rowlog[R + half * 8 + r] = __logf(rs[r]);
    }
}

// -------- Kernel 4: deterministic final reduction to scalar loss --------
__global__ void __launch_bounds__(256)
k_final(const float* __restrict__ rowlog, const float* __restrict__ pos,
        float* __restrict__ out) {
    __shared__ float sh[256];
    const int t = threadIdx.x;
    float sl = 0.f;
    for (int i = t; i < TWO_N; i += 256) sl += rowlog[i];
    float sp = 0.f;
    for (int i = t; i < N_ROWS; i += 256) sp += pos[i];
    sh[t] = sl - 2.0f * INV_TAU * sp;         // -(1/tau)*sum(pos_pairs) folded in
    __syncthreads();
    for (int s = 128; s > 0; s >>= 1) {
        if (t < s) sh[t] += sh[t + s];
        __syncthreads();
    }
    if (t == 0) out[0] = sh[0] / (float)TWO_N;
}

extern "C" void kernel_launch(void* const* d_in, const int* in_sizes, int n_in,
                              void* d_out, int out_size, void* d_ws, size_t ws_size,
                              hipStream_t stream) {
    const float* x1 = (const float*)d_in[0];
    const float* x2 = (const float*)d_in[1];
    char* ws = (char*)d_ws;
    _Float16* z      = (_Float16*)(ws + Z_OFF);
    float*    inv    = (float*)(ws + INV_OFF);
    float*    rowlog = (float*)(ws + ROWLOG_OFF);
    float*    pos    = (float*)(ws + POS_OFF);
    float*    out    = (float*)d_out;

    k_normalize  <<<TWO_N,      256, 0, stream>>>(x1, x2, z, inv);
    k_pos        <<<N_ROWS,     256, 0, stream>>>(x1, x2, inv, pos);
    k_gemm_rowsum<<<TWO_N/16/8, 256, 0, stream>>>(z, rowlog);
    k_final      <<<1,          256, 0, stream>>>(rowlog, pos, out);
}
